// FusedLlamaLowBitDecoderlayer_63385127354807
// MI455X (gfx1250) — compile-verified, matrix operand order checked by reading, not run on hardware
//
#include <hip/hip_runtime.h>

// ---------------------------------------------------------------------------
// Types for CDNA5 WMMA (wave32, v_wmma_f32_16x16x32_bf16) + TDM descriptors
// ---------------------------------------------------------------------------
typedef __attribute__((ext_vector_type(16))) __bf16 v16bf;
typedef __attribute__((ext_vector_type(8)))  __bf16 v8bf;
typedef __attribute__((ext_vector_type(4)))  __bf16 v4bf;
typedef __attribute__((ext_vector_type(8)))  float  v8f;
typedef __attribute__((ext_vector_type(4)))  unsigned int u32x4;
typedef __attribute__((ext_vector_type(8)))  int i32x8;
typedef __attribute__((ext_vector_type(4)))  int i32x4;

#define Bc   16
#define Tc   1024
#define Dc   512
#define Hc   2048
#define ROWS (Bc * Tc)   // 16384

union BF16x16 { v16bf v; v8bf h[2]; };

static __device__ __forceinline__ __bf16 f2bf(float f) {
  union { float f; unsigned u; } x; x.f = f;
  unsigned r = x.u + 0x7FFFu + ((x.u >> 16) & 1u);   // round-to-nearest-even
  unsigned short hs = (unsigned short)(r >> 16);
  return __builtin_bit_cast(__bf16, hs);
}
static __device__ __forceinline__ float bf2f(__bf16 h) {
  unsigned u = ((unsigned)__builtin_bit_cast(unsigned short, h)) << 16;
  return __builtin_bit_cast(float, u);
}

static __device__ __forceinline__ v8f wmma_bf16(v16bf a, v16bf b, v8f c) {
  return __builtin_amdgcn_wmma_f32_16x16x32_bf16(false, a, false, b,
                                                 (short)0, c, false, false);
}

// ---------------------------------------------------------------------------
// TDM: DMA one 128x32 bf16 tile (row stride K elems) into LDS, padded so the
// LDS row pitch is 40 halves (64B data + 16B pad).  D# per CDNA5 ISA ch.8.
// This toolchain exposes the 6-arg builtin:
//   (uint32x4 g0, int32x8 g1, int32x4 g2, int32x4 g3, int32x8 pad, i32 cpol)
// ---------------------------------------------------------------------------
static __device__ __forceinline__ void tdm_load_tile_128x32(
    const __bf16* gsrc, unsigned lds_addr, int K)
{
  unsigned long long ga = (unsigned long long)(uintptr_t)gsrc;
  u32x4 g0;
  g0[0] = 1u;                                          // count=1, user D#
  g0[1] = lds_addr;                                    // lds_addr[63:32]
  g0[2] = (unsigned)(ga & 0xFFFFFFFFu);                // global_addr lo
  g0[3] = (unsigned)((ga >> 32) & 0x01FFFFFFu) | (2u << 30);  // addr hi | type=2

  const unsigned td1 = 0x40000000u;                    // huge tensor_dim1 (no OOB)
  i32x8 g1;
  g1[0] = (int)((1u << 16)        // data_size = 2 bytes
              | (1u << 20)        // pad_enable
              | (3u << 22)        // pad_interval: 16 DWORDs (=64B)
              | (3u << 25));      // pad_amount:   4 DWORDs (=16B)
  g1[1] = (int)(((unsigned)K & 0xFFFFu) << 16);        // tensor_dim0 lo16
  g1[2] = (int)(((unsigned)K >> 16) | ((td1 & 0xFFFFu) << 16));
  g1[3] = (int)((td1 >> 16) | (32u << 16));            // tensor_dim1 hi | tile_dim0=32
  g1[4] = 128;                                         // tile_dim1=128, tile_dim2=0
  g1[5] = K;                                           // tensor_dim0_stride lo32
  g1[6] = 0;                                           // stride0 hi | stride1 lo
  g1[7] = 0;
  i32x4 z4; z4[0] = 0; z4[1] = 0; z4[2] = 0; z4[3] = 0;
  i32x8 z8; z8[0] = 0; z8[1] = 0; z8[2] = 0; z8[3] = 0;
  z8[4] = 0; z8[5] = 0; z8[6] = 0; z8[7] = 0;
  __builtin_amdgcn_tensor_load_to_lds(g0, g1, z4, z4, z8, 0);
}

// ---------------------------------------------------------------------------
// fp32 -> bf16 weight conversion (one-time per launch)
// ---------------------------------------------------------------------------
__global__ __launch_bounds__(256) void cvt_bf16_kernel(
    const float* __restrict__ in, __bf16* __restrict__ out)
{
  int i = blockIdx.x * 256 + threadIdx.x;
  out[i] = f2bf(in[i]);
}

// ---------------------------------------------------------------------------
// LayerNorm: one block (256 threads) per row of 512, fp32 in -> bf16 out
// ---------------------------------------------------------------------------
__global__ __launch_bounds__(256) void layernorm_kernel(
    const float* __restrict__ x, const float* __restrict__ w,
    const float* __restrict__ b, __bf16* __restrict__ y)
{
  const int row = blockIdx.x;
  const float* xr = x + (size_t)row * Dc;
  const int t = threadIdx.x;
  float v0 = xr[t], v1 = xr[t + 256];

  float s = v0 + v1;
  #pragma unroll
  for (int m = 16; m >= 1; m >>= 1) s += __shfl_xor(s, m, 32);
  __shared__ float red[8];
  const int wid = t >> 5, lid = t & 31;
  if (lid == 0) red[wid] = s;
  __syncthreads();
  float tot = 0.f;
  #pragma unroll
  for (int i = 0; i < 8; i++) tot += red[i];
  const float mean = tot * (1.0f / Dc);

  float d0 = v0 - mean, d1 = v1 - mean;
  float q = d0 * d0 + d1 * d1;
  #pragma unroll
  for (int m = 16; m >= 1; m >>= 1) q += __shfl_xor(q, m, 32);
  __shared__ float red2[8];
  if (lid == 0) red2[wid] = q;
  __syncthreads();
  float vt = 0.f;
  #pragma unroll
  for (int i = 0; i < 8; i++) vt += red2[i];
  const float rstd = rsqrtf(vt * (1.0f / Dc) + 1e-12f);

  __bf16* yr = y + (size_t)row * Dc;
  yr[t]       = f2bf(d0 * rstd * w[t]       + b[t]);
  yr[t + 256] = f2bf(d1 * rstd * w[t + 256] + b[t + 256]);
}

// ---------------------------------------------------------------------------
// bf16 WMMA GEMM with TDM double-buffered staging:
//   C[M,N] = A[M,K](bf16) * W[N,K](bf16)^T + epilogue
// block = 256 threads (8 waves), tile 128x128, k-step 32.
// EPI: 0 = +bias -> bf16 ; 1 = +bias+add0+add1 -> f32 ;
//      2 = +bias,relu -> bf16 ; 3 = +bias+add0 -> f32
// ---------------------------------------------------------------------------
template <int EPI>
__global__ __launch_bounds__(256) void gemm_bf16_kernel(
    const __bf16* __restrict__ A, const __bf16* __restrict__ W,
    const float* __restrict__ bias, const float* __restrict__ add0,
    const float* __restrict__ add1, void* __restrict__ C,
    int M, int N, int K)
{
  __shared__ __align__(16) __bf16 At[2][128 * 40];
  __shared__ __align__(16) __bf16 Wt[2][128 * 40];

  const int nbase = blockIdx.x * 128;
  const int mbase = blockIdx.y * 128;
  const int tid  = threadIdx.x;
  const int lane = tid & 31;
  const int wave = tid >> 5;
  const int wm = wave & 1;          // 64-row half of the tile
  const int wn = wave >> 1;         // 32-col slice of the tile
  const int lrow = lane & 15;
  const int koff = (lane < 16) ? 0 : 8;   // K-chunk select per frag layout

  const __bf16* Ag = A + (size_t)mbase * K;
  const __bf16* Wg = W + (size_t)nbase * K;

  v8f acc[4][2];
  #pragma unroll
  for (int i = 0; i < 4; i++)
    #pragma unroll
    for (int j = 0; j < 2; j++) acc[i][j] = (v8f)0.f;

  // prologue: DMA first k-slice into buffer 0 (wave 0 drives the TDM)
  if (wave == 0) {
    tdm_load_tile_128x32(Ag, (unsigned)(uintptr_t)&At[0][0], K);
    tdm_load_tile_128x32(Wg, (unsigned)(uintptr_t)&Wt[0][0], K);
  }

  int cur = 0;
  for (int k0 = 0; k0 < K; k0 += 32) {
    __syncthreads();   // everyone done reading buffer cur^1 (prev iter)
    if (wave == 0) {
      if (k0 + 32 < K) {
        tdm_load_tile_128x32(Ag + (k0 + 32), (unsigned)(uintptr_t)&At[cur ^ 1][0], K);
        tdm_load_tile_128x32(Wg + (k0 + 32), (unsigned)(uintptr_t)&Wt[cur ^ 1][0], K);
        __builtin_amdgcn_s_wait_tensorcnt(2);  // current pair landed
      } else {
        __builtin_amdgcn_s_wait_tensorcnt(0);
      }
    }
    __syncthreads();   // tile `cur` visible to all waves

    const __bf16* Ab = At[cur];
    const __bf16* Wb = Wt[cur];
    BF16x16 af[4], bfr[2];
    #pragma unroll
    for (int i = 0; i < 4; i++) {
      int r = wm * 64 + i * 16 + lrow;
      af[i].h[0] = *(const v8bf*)&Ab[r * 40 + koff];
      af[i].h[1] = *(const v8bf*)&Ab[r * 40 + koff + 16];
    }
    #pragma unroll
    for (int j = 0; j < 2; j++) {
      int r = wn * 32 + j * 16 + lrow;
      bfr[j].h[0] = *(const v8bf*)&Wb[r * 40 + koff];
      bfr[j].h[1] = *(const v8bf*)&Wb[r * 40 + koff + 16];
    }
    #pragma unroll
    for (int i = 0; i < 4; i++)
      #pragma unroll
      for (int j = 0; j < 2; j++)
        acc[i][j] = wmma_bf16(af[i].v, bfr[j].v, acc[i][j]);
    cur ^= 1;
  }

  const int rowoff = (lane >= 16) ? 8 : 0;
  #pragma unroll
  for (int i = 0; i < 4; i++) {
    #pragma unroll
    for (int j = 0; j < 2; j++) {
      int col = nbase + wn * 32 + j * 16 + lrow;
      float bv = bias[col];
      #pragma unroll
      for (int jj = 0; jj < 8; jj++) {
        int row = mbase + wm * 64 + i * 16 + jj + rowoff;
        size_t idx = (size_t)row * N + col;
        float v = acc[i][j][jj] + bv;
        if (EPI == 1) v += add0[idx] + add1[idx];
        if (EPI == 2) v = fmaxf(v, 0.f);
        if (EPI == 3) v += add0[idx];
        if (EPI == 0 || EPI == 2) ((__bf16*)C)[idx] = f2bf(v);
        else                      ((float*)C)[idx]  = v;
      }
    }
  }
}

// ---------------------------------------------------------------------------
// FSMN depthwise conv (k=11, pad 5/5): out = (conv(v*m) + v*m) * m  (fp32 out)
// v lives in bf16 qkv buffer at column offset 2*Dc. One block per (b,t).
// ---------------------------------------------------------------------------
__global__ __launch_bounds__(256) void fsmn_kernel(
    const __bf16* __restrict__ qkv, const float* __restrict__ masks,
    const float* __restrict__ fw, float* __restrict__ out)
{
  const int bt = blockIdx.x;
  const int b = bt / Tc;
  const int t = bt % Tc;
  const float mt = masks[bt];

  #pragma unroll
  for (int e = 0; e < 2; e++) {
    const int d = threadIdx.x + e * 256;
    float r = 0.f;
    if (mt > 0.f) {
      float vc = bf2f(qkv[(size_t)bt * 1536 + 1024 + d]);
      float s = 0.f;
      #pragma unroll
      for (int j = 0; j < 11; j++) {
        int tt = t + j - 5;
        if (tt >= 0 && tt < Tc) {
          float mm = masks[b * Tc + tt];
          float vv = bf2f(qkv[((size_t)(b * Tc + tt)) * 1536 + 1024 + d]);
          s += vv * mm * fw[d * 11 + j];
        }
      }
      r = (s + vc * mt) * mt;
    }
    out[(size_t)bt * Dc + d] = r;
  }
}

// ---------------------------------------------------------------------------
// Flash attention (4 heads, dk=128). Block = 8 waves, owns (b, h, 128 q rows);
// each wave owns 16 q rows. Online softmax over 32-key blocks. bf16 in/out.
// ---------------------------------------------------------------------------
__global__ __launch_bounds__(256) void attn_kernel(
    const __bf16* __restrict__ qkv, const float* __restrict__ masks,
    __bf16* __restrict__ ctx)
{
  const int blk = blockIdx.x;           // B*4*(T/128) = 512
  const int qt = blk & 7;
  const int h  = (blk >> 3) & 3;
  const int b  = blk >> 5;
  const int tid  = threadIdx.x;
  const int lane = tid & 31;
  const int wave = tid >> 5;
  const int lrow = lane & 15;
  const int koff   = (lane < 16) ? 0 : 8;
  const int rowoff = (lane < 16) ? 0 : 8;

  __shared__ __align__(16) __bf16 Kt[32 * 136];
  __shared__ __align__(16) __bf16 Vt[128 * 40];     // transposed: [d][k]
  __shared__ __align__(16) __bf16 Pt[8 * 16 * 40];  // per-wave P patch

  // Q fragments: direct bf16 b128 loads, no conversion
  const int qrow = qt * 128 + wave * 16 + lrow;
  const __bf16* qp = qkv + ((size_t)(b * Tc) + qrow) * 1536 + h * 128;
  BF16x16 qf[4];
  #pragma unroll
  for (int c = 0; c < 4; c++) {
    const __bf16* p = qp + c * 32 + koff;
    qf[c].h[0] = *(const v8bf*)p;
    qf[c].h[1] = *(const v8bf*)(p + 16);
  }

  v8f O[8];
  #pragma unroll
  for (int n = 0; n < 8; n++) O[n] = (v8f)0.f;
  float mrun[8], lrun[8];
  #pragma unroll
  for (int jj = 0; jj < 8; jj++) { mrun[jj] = -1e30f; lrun[jj] = 0.f; }

  const float scale = 0.08838834764831845f;  // 1/sqrt(128)

  for (int kb = 0; kb < Tc; kb += 32) {
    __syncthreads();
    {   // cooperative K-tile (row major) and V^T-tile (transposed) staging
      const int tk = tid >> 3;      // 0..31
      const int d0 = (tid & 7) << 4;
      const __bf16* kp = qkv + ((size_t)(b * Tc) + kb + tk) * 1536 + 512  + h * 128 + d0;
      const __bf16* vp = qkv + ((size_t)(b * Tc) + kb + tk) * 1536 + 1024 + h * 128 + d0;
      *(v8bf*)&Kt[tk * 136 + d0]     = *(const v8bf*)kp;
      *(v8bf*)&Kt[tk * 136 + d0 + 8] = *(const v8bf*)(kp + 8);
      v8bf v0 = *(const v8bf*)vp;
      v8bf v1 = *(const v8bf*)(vp + 8);
      #pragma unroll
      for (int i = 0; i < 8; i++) {
        Vt[(d0 + i) * 40 + tk]     = v0[i];
        Vt[(d0 + 8 + i) * 40 + tk] = v1[i];
      }
    }
    __syncthreads();

    // S(16x32) = Q(16x128) * K(32x128)^T : two 16x16 tiles, 4 WMMA each
    v8f s[2];
    s[0] = (v8f)0.f; s[1] = (v8f)0.f;
    #pragma unroll
    for (int nt = 0; nt < 2; nt++) {
      const int krow = nt * 16 + lrow;
      #pragma unroll
      for (int c = 0; c < 4; c++) {
        BF16x16 kf;
        const int off = c * 32 + koff;
        kf.h[0] = *(const v8bf*)&Kt[krow * 136 + off];
        kf.h[1] = *(const v8bf*)&Kt[krow * 136 + off + 16];
        s[nt] = wmma_bf16(qf[c].v, kf.v, s[nt]);
      }
    }

    const float mk0 = masks[b * Tc + kb + lrow];
    const float mk1 = masks[b * Tc + kb + 16 + lrow];

    float p0s[8], p1s[8];
    #pragma unroll
    for (int jj = 0; jj < 8; jj++) {
      float s0 = s[0][jj] * scale; if (mk0 <= 0.f) s0 = -1e30f;
      float s1 = s[1][jj] * scale; if (mk1 <= 0.f) s1 = -1e30f;
      float mx = fmaxf(s0, s1);
      #pragma unroll
      for (int m = 1; m < 16; m <<= 1) mx = fmaxf(mx, __shfl_xor(mx, m, 32));
      const float mnew = fmaxf(mrun[jj], mx);
      const float sc = __expf(mrun[jj] - mnew);
      float p0 = __expf(s0 - mnew);
      float p1 = __expf(s1 - mnew);
      if (s0 <= -1e29f) p0 = 0.f;
      if (s1 <= -1e29f) p1 = 0.f;
      float rs = p0 + p1;
      #pragma unroll
      for (int m = 1; m < 16; m <<= 1) rs += __shfl_xor(rs, m, 32);
      lrun[jj] = lrun[jj] * sc + rs;
      mrun[jj] = mnew;
      #pragma unroll
      for (int n = 0; n < 8; n++) O[n][jj] *= sc;
      p0s[jj] = p0; p1s[jj] = p1;
    }

    // C-layout -> A-frag layout for P via wave-private LDS patch
    __bf16* P = &Pt[wave * 16 * 40];
    #pragma unroll
    for (int jj = 0; jj < 8; jj++) {
      const int r = jj + rowoff;
      P[r * 40 + lrow]      = f2bf(p0s[jj]);
      P[r * 40 + 16 + lrow] = f2bf(p1s[jj]);
    }
    BF16x16 pa;
    pa.h[0] = *(const v8bf*)&P[lrow * 40 + koff];
    pa.h[1] = *(const v8bf*)&P[lrow * 40 + koff + 16];

    // O(16x128) += P(16x32) * V(32x128): 8 WMMA against V^T B-frags
    #pragma unroll
    for (int n = 0; n < 8; n++) {
      const int vr = n * 16 + lrow;
      BF16x16 vf;
      vf.h[0] = *(const v8bf*)&Vt[vr * 40 + koff];
      vf.h[1] = *(const v8bf*)&Vt[vr * 40 + koff + 16];
      O[n] = wmma_bf16(pa.v, vf.v, O[n]);
    }
  }

  float inv[8];
  #pragma unroll
  for (int jj = 0; jj < 8; jj++) inv[jj] = (lrun[jj] > 0.f) ? 1.f / lrun[jj] : 0.f;

  #pragma unroll
  for (int n = 0; n < 8; n++) {
    #pragma unroll
    for (int jj = 0; jj < 8; jj++) {
      const int row = qt * 128 + wave * 16 + jj + rowoff;
      ctx[((size_t)(b * Tc) + row) * Dc + h * 128 + n * 16 + lrow] = f2bf(O[n][jj] * inv[jj]);
    }
  }
}

// ---------------------------------------------------------------------------
// Orchestration
// ---------------------------------------------------------------------------
extern "C" void kernel_launch(void* const* d_in, const int* in_sizes, int n_in,
                              void* d_out, int out_size, void* d_ws, size_t ws_size,
                              hipStream_t stream) {
  const float* x      = (const float*)d_in[0];
  const float* masks  = (const float*)d_in[1];
  const float* ln0_w  = (const float*)d_in[2];
  const float* ln0_b  = (const float*)d_in[3];
  const float* ln1_w  = (const float*)d_in[4];
  const float* ln1_b  = (const float*)d_in[5];
  const float* fsmn_w = (const float*)d_in[6];
  const float* qkv_w  = (const float*)d_in[7];
  const float* qkv_b  = (const float*)d_in[8];
  const float* out_w  = (const float*)d_in[9];
  const float* out_b  = (const float*)d_in[10];
  const float* w1_w   = (const float*)d_in[11];
  const float* w1_b   = (const float*)d_in[12];
  const float* w2_w   = (const float*)d_in[13];
  const float* w2_b   = (const float*)d_in[14];
  float* out = (float*)d_out;

  // workspace layout
  const size_t R = (size_t)ROWS;
  char* ws = (char*)d_ws;
  float*  fsmn  = (float*)ws;                 ws += R * 512 * 4;
  float*  x1    = (float*)ws;                 ws += R * 512 * 4;
  __bf16* hB    = (__bf16*)ws;                ws += R * 512 * 2;   // LN out (reused)
  __bf16* qkvB  = (__bf16*)ws;                ws += R * 1536 * 2;
  __bf16* ctxB  = (__bf16*)ws;                ws += R * 512 * 2;
  __bf16* hhB   = (__bf16*)ws;                ws += R * 2048 * 2;
  __bf16* qkvWb = (__bf16*)ws;                ws += (size_t)1536 * 512 * 2;
  __bf16* outWb = (__bf16*)ws;                ws += (size_t)512 * 512 * 2;
  __bf16* w1Wb  = (__bf16*)ws;                ws += (size_t)2048 * 512 * 2;
  __bf16* w2Wb  = (__bf16*)ws;                ws += (size_t)512 * 2048 * 2;

  // 0) one-time weight conversion fp32 -> bf16
  cvt_bf16_kernel<<<(1536 * 512) / 256, 256, 0, stream>>>(qkv_w, qkvWb);
  cvt_bf16_kernel<<<(512 * 512) / 256, 256, 0, stream>>>(out_w, outWb);
  cvt_bf16_kernel<<<(2048 * 512) / 256, 256, 0, stream>>>(w1_w, w1Wb);
  cvt_bf16_kernel<<<(512 * 2048) / 256, 256, 0, stream>>>(w2_w, w2Wb);

  // 1) LN0 -> hB (bf16)
  layernorm_kernel<<<ROWS, 256, 0, stream>>>(x, ln0_w, ln0_b, hB);
  // 2) qkvB = hB @ qkv_w^T + qkv_b   (bf16 out)
  gemm_bf16_kernel<0><<<dim3(1536 / 128, ROWS / 128), 256, 0, stream>>>(
      hB, qkvWb, qkv_b, nullptr, nullptr, qkvB, ROWS, 1536, 512);
  // 3) FSMN memory branch (fp32 out)
  fsmn_kernel<<<ROWS, 256, 0, stream>>>(qkvB, masks, fsmn_w, fsmn);
  // 4) Flash attention -> ctxB (bf16)
  attn_kernel<<<Bc * 4 * (Tc / 128), 256, 0, stream>>>(qkvB, masks, ctxB);
  // 5) x1 = ctxB @ out_w^T + out_b + fsmn + x   (fp32 out)
  gemm_bf16_kernel<1><<<dim3(512 / 128, ROWS / 128), 256, 0, stream>>>(
      ctxB, outWb, out_b, fsmn, x, x1, ROWS, 512, 512);
  // 6) LN1 -> hB (bf16, reuse)
  layernorm_kernel<<<ROWS, 256, 0, stream>>>(x1, ln1_w, ln1_b, hB);
  // 7) hhB = relu(hB @ w1_w^T + w1_b)   (bf16 out)
  gemm_bf16_kernel<2><<<dim3(2048 / 128, ROWS / 128), 256, 0, stream>>>(
      hB, w1Wb, w1_b, nullptr, nullptr, hhB, ROWS, 2048, 512);
  // 8) out = hhB @ w2_w^T + w2_b + x1   (fp32 out)
  gemm_bf16_kernel<3><<<dim3(512 / 128, ROWS / 128), 256, 0, stream>>>(
      hhB, w2Wb, w2_b, x1, nullptr, out, ROWS, 512, 2048);
}
